// MultiHeadAttentionLayer_5145370821268
// MI455X (gfx1250) — matrix operand order repaired and checked by reading, not compiled, not run on hardware
//
#include <hip/hip_runtime.h>
#include <hip/hip_bf16.h>

// ---------------------------------------------------------------------------
// MI455X (gfx1250) multi-head attention, bf16 WMMA path, flash-style softmax.
// B=4, S=2048, D_MODEL=1024, H=16, DK=64.
// ---------------------------------------------------------------------------

typedef __attribute__((ext_vector_type(16))) __bf16 bf16x16;
typedef __attribute__((ext_vector_type(8)))  __bf16 bf16x8;
typedef __attribute__((ext_vector_type(4)))  __bf16 bf16x4;
typedef __attribute__((ext_vector_type(8)))  float  f32x8;

#define SEQ    2048
#define DMODEL 1024
#define NHEAD  16
#define DK     64

// LDS row stride in halves for a 64-wide K chunk: 72 halves = 144 B
// (multiple of 16 B for b128 fragment reads, non-power-of-2 to spread banks).
#define KSTR 72

__device__ __forceinline__ bf16x16 cat8(bf16x8 lo, bf16x8 hi) {
    return __builtin_shufflevector(lo, hi, 0, 1, 2, 3, 4, 5, 6, 7,
                                           8, 9, 10, 11, 12, 13, 14, 15);
}

// ---------------------------------------------------------------------------
// Kernel 1: projection GEMM.  X[f32, 8192x1024] @ W[f32, 1024x1024] + bias,
// output bf16 written head-partitioned as [B][H][S][DK].
// Block: 256 thr (8 waves). Block tile 128(M) x 64(N). K chunk = 64
// (two 32-deep WMMA sub-steps per stage -> 8 WMMAs/wave between barriers).
// ---------------------------------------------------------------------------
__global__ __launch_bounds__(256)
void mha_qkv_proj_kernel(const float* __restrict__ X,
                         const float* __restrict__ W,
                         const float* __restrict__ bias,
                         __bf16* __restrict__ dst)
{
    __shared__ __align__(16) __bf16 ldsA[128 * KSTR];  // [row][k]
    __shared__ __align__(16) __bf16 ldsB[64 * KSTR];   // [n][k] transposed

    const int t  = threadIdx.x;
    const int w  = t >> 5;
    const int l  = t & 31;
    const int lh = l >> 4;   // lane half (0/1)
    const int ll = l & 15;
    const int m0 = blockIdx.x * 128;
    const int n0 = blockIdx.y * 64;

    f32x8 c[4] = {};

    for (int k0 = 0; k0 < DMODEL; k0 += 64) {
        __syncthreads();
        // Prefetch next-stage A/B lines into L2/WGP$ (global_prefetch_b8).
        if (k0 + 64 < DMODEL) {
            __builtin_prefetch(&X[(size_t)(m0 + (t >> 1)) * DMODEL + k0 + 64 + (t & 1) * 32], 0, 3);
            __builtin_prefetch(&W[(size_t)(k0 + 64 + (t >> 2)) * DMODEL + n0 + (t & 3) * 16], 0, 3);
        }
        // Stage A tile: 128x64 f32 -> bf16 LDS. 2048 float4 loads / 256 thr.
#pragma unroll
        for (int i = 0; i < 8; ++i) {
            int lin = t + i * 256;            // 0..2047
            int row = lin >> 4;               // 16 float4 per row
            int kq  = (lin & 15) << 2;
            const float4 v = *(const float4*)&X[(size_t)(m0 + row) * DMODEL + k0 + kq];
            bf16x4 hv;
            hv[0] = (__bf16)v.x; hv[1] = (__bf16)v.y;
            hv[2] = (__bf16)v.z; hv[3] = (__bf16)v.w;
            *(bf16x4*)&ldsA[row * KSTR + kq] = hv;
        }
        // Stage B tile transposed: W[k0+kk][n0+nn] -> ldsB[nn][kk].
#pragma unroll
        for (int i = 0; i < 4; ++i) {
            int lin = t + i * 256;            // 0..1023
            int kk  = lin >> 4;               // 0..63
            int nq  = (lin & 15) << 2;
            const float4 v = *(const float4*)&W[(size_t)(k0 + kk) * DMODEL + n0 + nq];
            ldsB[(nq + 0) * KSTR + kk] = (__bf16)v.x;
            ldsB[(nq + 1) * KSTR + kk] = (__bf16)v.y;
            ldsB[(nq + 2) * KSTR + kk] = (__bf16)v.z;
            ldsB[(nq + 3) * KSTR + kk] = (__bf16)v.w;
        }
        __syncthreads();

        const int arow = w * 16 + ll;
#pragma unroll
        for (int sub = 0; sub < 2; ++sub) {
            const int kb = sub * 32;
            // A fragment: row = arow; e<8: k=kb+lh*8+e ; e>=8: k=kb+16+lh*8+(e-8)
            bf16x8 a_lo = *(const bf16x8*)&ldsA[arow * KSTR + kb + lh * 8];
            bf16x8 a_hi = *(const bf16x8*)&ldsA[arow * KSTR + kb + 16 + lh * 8];
            bf16x16 a = cat8(a_lo, a_hi);
#pragma unroll
            for (int j = 0; j < 4; ++j) {
                // B fragment: col = j*16+ll; e<8: k=kb+lh*8+e ; e>=8: k=kb+16+lh*8+(e-8)
                const int bcol = j * 16 + ll;
                bf16x8 b_lo = *(const bf16x8*)&ldsB[bcol * KSTR + kb + lh * 8];
                bf16x8 b_hi = *(const bf16x8*)&ldsB[bcol * KSTR + kb + 16 + lh * 8];
                bf16x16 bb = cat8(b_lo, b_hi);
                c[j] = __builtin_amdgcn_wmma_f32_16x16x32_bf16(
                    false, a, false, bb, (short)0, c[j], false, false);
            }
        }
    }

    // Epilogue: C/D layout row = r + 8*lh, col = ll. Add bias, store bf16
    // into [B][H][S][DK] head-partitioned layout.
#pragma unroll
    for (int j = 0; j < 4; ++j) {
#pragma unroll
        for (int r = 0; r < 8; ++r) {
            const int M = m0 + w * 16 + r + lh * 8;
            const int N = n0 + j * 16 + ll;
            const float v = c[j][r] + bias[N];
            const int b = M >> 11;            // / SEQ
            const int s = M & (SEQ - 1);
            const int h = N >> 6;             // / DK
            const int d = N & (DK - 1);
            dst[(((size_t)(b * NHEAD + h) * SEQ + s) << 6) + d] = (__bf16)v;
        }
    }
}

// ---------------------------------------------------------------------------
// Kernel 2: flash attention over bf16 Q/K/V in [B][H][S][DK] layout.
// Grid: (S/128, B*H). Block 256 thr = 8 waves; each wave owns 16 query rows.
// Key loop in chunks of 32: 4 score WMMAs + online softmax + 4 P@V WMMAs.
// ---------------------------------------------------------------------------
__global__ __launch_bounds__(256)
void mha_flash_attn_kernel(const __bf16* __restrict__ Qb,
                           const __bf16* __restrict__ Kb,
                           const __bf16* __restrict__ Vb,
                           const int*    __restrict__ mask,
                           __bf16* __restrict__ Ob)
{
    __shared__ __align__(16) __bf16 ldsVt[64 * 40];       // V tile transposed [d][k]
    __shared__ __align__(16) __bf16 ldsP[8 * 16 * 40];    // per-wave P re-layout

    const int t  = threadIdx.x;
    const int w  = t >> 5;
    const int l  = t & 31;
    const int lh = l >> 4;
    const int ll = l & 15;
    const int bh = blockIdx.y;              // b*H + h
    const int b  = bh >> 4;
    const int h  = bh & (NHEAD - 1);
    const int q0 = blockIdx.x * 128;
    const int qw = q0 + w * 16;             // this wave's 16 query rows
    const size_t headBase = (size_t)bh * SEQ * DK;

    // Q A-fragments for contraction halves d=[0,32) and [32,64); loaded once.
    const __bf16* qrow = Qb + headBase + (size_t)(qw + ll) * DK;
    bf16x16 aQ[2];
#pragma unroll
    for (int dh = 0; dh < 2; ++dh) {
        bf16x8 lo = *(const bf16x8*)&qrow[dh * 32 + lh * 8];
        bf16x8 hi = *(const bf16x8*)&qrow[dh * 32 + 16 + lh * 8];
        aQ[dh] = cat8(lo, hi);
    }

    f32x8 o[4] = {};
    float mrow[8], lrow[8];
#pragma unroll
    for (int r = 0; r < 8; ++r) { mrow[r] = -3.0e38f; lrow[r] = 0.0f; }

    for (int kt = 0; kt < SEQ; kt += 32) {
        // Prefetch next key tile's K/V rows while this tile is computed.
        if (kt + 32 < SEQ) {
            __builtin_prefetch(&Kb[headBase + (size_t)(kt + 32 + (t & 31)) * DK], 0, 3);
            __builtin_prefetch(&Vb[headBase + (size_t)(kt + 32 + (t & 31)) * DK], 0, 3);
        }
        __syncthreads();
        // Stage V tile (32 keys x 64 d) transposed into LDS.
        {
            const int kr = t >> 3;          // 0..31
            const int dq = (t & 7) << 3;    // 0..56
            bf16x8 v = *(const bf16x8*)&Vb[headBase + (size_t)(kt + kr) * DK + dq];
#pragma unroll
            for (int i = 0; i < 8; ++i) ldsVt[(dq + i) * 40 + kr] = v[i];
        }
        __syncthreads();

        // Scores: two 16x16 key sub-tiles, contraction over d=64 (2 WMMAs each).
        f32x8 sc[2];
#pragma unroll
        for (int s = 0; s < 2; ++s) {
            const __bf16* krow = Kb + headBase + (size_t)(kt + s * 16 + ll) * DK;
            f32x8 cz = {};
#pragma unroll
            for (int dh = 0; dh < 2; ++dh) {
                bf16x8 lo = *(const bf16x8*)&krow[dh * 32 + lh * 8];
                bf16x8 hi = *(const bf16x8*)&krow[dh * 32 + 16 + lh * 8];
                bf16x16 bk = cat8(lo, hi);
                cz = __builtin_amdgcn_wmma_f32_16x16x32_bf16(
                    false, aQ[dh], false, bk, (short)0, cz, false, false);
            }
            sc[s] = cz;
        }

        // Mask + online softmax, per C/D row (r + 8*lh), cols across 16 lanes.
#pragma unroll
        for (int r = 0; r < 8; ++r) {
            const int q = qw + r + lh * 8;
            const int* mp = &mask[((size_t)(b * SEQ + q) << 11) + kt];
            float x0 = sc[0][r] * 0.125f;          // 1/sqrt(64)
            float x1 = sc[1][r] * 0.125f;
            if (mp[ll] == 0)      x0 = -1.0e9f;
            if (mp[16 + ll] == 0) x1 = -1.0e9f;

            float mx = fmaxf(x0, x1);
            mx = fmaxf(mx, __shfl_xor(mx, 1));
            mx = fmaxf(mx, __shfl_xor(mx, 2));
            mx = fmaxf(mx, __shfl_xor(mx, 4));
            mx = fmaxf(mx, __shfl_xor(mx, 8));

            const float newm  = fmaxf(mrow[r], mx);
            const float alpha = __expf(mrow[r] - newm);
            const float p0 = __expf(x0 - newm);
            const float p1 = __expf(x1 - newm);
            float ps = p0 + p1;
            ps += __shfl_xor(ps, 1);
            ps += __shfl_xor(ps, 2);
            ps += __shfl_xor(ps, 4);
            ps += __shfl_xor(ps, 8);

            lrow[r] = lrow[r] * alpha + ps;
            mrow[r] = newm;
#pragma unroll
            for (int j = 0; j < 4; ++j) o[j][r] *= alpha;

            // Scatter P (C-layout) into wave-private LDS as [row][key].
            const int prow = r + lh * 8;
            ldsP[(w * 16 + prow) * 40 + ll]      = (__bf16)p0;
            ldsP[(w * 16 + prow) * 40 + 16 + ll] = (__bf16)p1;
        }

        // Re-load P as 16x32 A-fragment (same-wave LDS is in-order).
        bf16x8 plo = *(const bf16x8*)&ldsP[(w * 16 + ll) * 40 + lh * 8];
        bf16x8 phi = *(const bf16x8*)&ldsP[(w * 16 + ll) * 40 + 16 + lh * 8];
        bf16x16 aP = cat8(plo, phi);

        // P @ V: four 16-wide d sub-tiles.
#pragma unroll
        for (int j = 0; j < 4; ++j) {
            const int dcol = j * 16 + ll;
            bf16x8 vlo = *(const bf16x8*)&ldsVt[dcol * 40 + lh * 8];
            bf16x8 vhi = *(const bf16x8*)&ldsVt[dcol * 40 + 16 + lh * 8];
            bf16x16 bv = cat8(vlo, vhi);
            o[j] = __builtin_amdgcn_wmma_f32_16x16x32_bf16(
                false, aP, false, bv, (short)0, o[j], false, false);
        }
    }

    // Normalize and store bf16 to [B][S][H*DK] (= row-major 8192x1024).
#pragma unroll
    for (int j = 0; j < 4; ++j) {
#pragma unroll
        for (int r = 0; r < 8; ++r) {
            const int q = qw + r + lh * 8;
            const int d = j * 16 + ll;
            const float v = o[j][r] / lrow[r];
            Ob[((size_t)(b * SEQ + q) << 10) + h * DK + d] = (__bf16)v;
        }
    }
}

// ---------------------------------------------------------------------------
// Kernel 3: output projection. A[bf16, 8192x1024] @ Wo[f32] + bo -> f32 out.
// Same tiling as kernel 1 (K chunk 64).
// ---------------------------------------------------------------------------
__global__ __launch_bounds__(256)
void mha_out_proj_kernel(const __bf16* __restrict__ A,
                         const float* __restrict__ W,
                         const float* __restrict__ bias,
                         float* __restrict__ out)
{
    __shared__ __align__(16) __bf16 ldsA[128 * KSTR];
    __shared__ __align__(16) __bf16 ldsB[64 * KSTR];

    const int t  = threadIdx.x;
    const int w  = t >> 5;
    const int l  = t & 31;
    const int lh = l >> 4;
    const int ll = l & 15;
    const int m0 = blockIdx.x * 128;
    const int n0 = blockIdx.y * 64;

    f32x8 c[4] = {};

    for (int k0 = 0; k0 < DMODEL; k0 += 64) {
        __syncthreads();
        if (k0 + 64 < DMODEL) {
            __builtin_prefetch(&A[(size_t)(m0 + (t >> 1)) * DMODEL + k0 + 64 + (t & 1) * 32], 0, 3);
            __builtin_prefetch(&W[(size_t)(k0 + 64 + (t >> 2)) * DMODEL + n0 + (t & 3) * 16], 0, 3);
        }
        // Stage A tile (already bf16): 128x64 halves, 1024 x b128 / 256 thr.
#pragma unroll
        for (int i = 0; i < 4; ++i) {
            int lin = t + i * 256;            // 0..1023
            int row = lin >> 3;               // 8 chunks of 8 per row
            int kq  = (lin & 7) << 3;
            bf16x8 v = *(const bf16x8*)&A[(size_t)(m0 + row) * DMODEL + k0 + kq];
            *(bf16x8*)&ldsA[row * KSTR + kq] = v;
        }
#pragma unroll
        for (int i = 0; i < 4; ++i) {
            int lin = t + i * 256;            // 0..1023
            int kk  = lin >> 4;               // 0..63
            int nq  = (lin & 15) << 2;
            const float4 v = *(const float4*)&W[(size_t)(k0 + kk) * DMODEL + n0 + nq];
            ldsB[(nq + 0) * KSTR + kk] = (__bf16)v.x;
            ldsB[(nq + 1) * KSTR + kk] = (__bf16)v.y;
            ldsB[(nq + 2) * KSTR + kk] = (__bf16)v.z;
            ldsB[(nq + 3) * KSTR + kk] = (__bf16)v.w;
        }
        __syncthreads();

        const int arow = w * 16 + ll;
#pragma unroll
        for (int sub = 0; sub < 2; ++sub) {
            const int kb = sub * 32;
            bf16x8 a_lo = *(const bf16x8*)&ldsA[arow * KSTR + kb + lh * 8];
            bf16x8 a_hi = *(const bf16x8*)&ldsA[arow * KSTR + kb + 16 + lh * 8];
            bf16x16 a = cat8(a_lo, a_hi);
#pragma unroll
            for (int j = 0; j < 4; ++j) {
                const int bcol = j * 16 + ll;
                bf16x8 b_lo = *(const bf16x8*)&ldsB[bcol * KSTR + kb + lh * 8];
                bf16x8 b_hi = *(const bf16x8*)&ldsB[bcol * KSTR + kb + 16 + lh * 8];
                bf16x16 bb = cat8(b_lo, b_hi);
                c[j] = __builtin_amdgcn_wmma_f32_16x16x32_bf16(
                    false, a, false, bb, (short)0, c[j], false, false);
            }
        }
    }

#pragma unroll
    for (int j = 0; j < 4; ++j) {
#pragma unroll
        for (int r = 0; r < 8; ++r) {
            const int M = m0 + w * 16 + r + lh * 8;
            const int N = n0 + j * 16 + ll;
            out[(size_t)M * DMODEL + N] = c[j][r] + bias[N];
        }
    }
}

// ---------------------------------------------------------------------------
// Launch: 3x projection, 1x flash attention, 1x output projection.
// Workspace: Q,K,V,O bf16 head tensors, 16 MiB each = 64 MiB.
// ---------------------------------------------------------------------------
extern "C" void kernel_launch(void* const* d_in, const int* in_sizes, int n_in,
                              void* d_out, int out_size, void* d_ws, size_t ws_size,
                              hipStream_t stream)
{
    (void)in_sizes; (void)n_in; (void)out_size; (void)ws_size;

    const float* Xq   = (const float*)d_in[0];
    const float* Xk   = (const float*)d_in[1];
    const float* Xv   = (const float*)d_in[2];
    const int*   mask = (const int*)  d_in[3];
    const float* Wq   = (const float*)d_in[4];
    const float* bq   = (const float*)d_in[5];
    const float* Wk   = (const float*)d_in[6];
    const float* bk   = (const float*)d_in[7];
    const float* Wv   = (const float*)d_in[8];
    const float* bv   = (const float*)d_in[9];
    const float* Wo   = (const float*)d_in[10];
    const float* bo   = (const float*)d_in[11];
    float* out = (float*)d_out;

    const size_t tensorElems = (size_t)4 * NHEAD * SEQ * DK;   // 8 Mi elems
    __bf16* Qb = (__bf16*)d_ws;
    __bf16* Kb = Qb + tensorElems;
    __bf16* Vb = Kb + tensorElems;
    __bf16* Ob = Vb + tensorElems;

    dim3 gProj(64, 16);   // 8192/128 x 1024/64
    mha_qkv_proj_kernel<<<gProj, 256, 0, stream>>>(Xq, Wq, bq, Qb);
    mha_qkv_proj_kernel<<<gProj, 256, 0, stream>>>(Xk, Wk, bk, Kb);
    mha_qkv_proj_kernel<<<gProj, 256, 0, stream>>>(Xv, Wv, bv, Vb);

    dim3 gAttn(16, 64);   // 2048/128 q-tiles x (B*H)
    mha_flash_attn_kernel<<<gAttn, 256, 0, stream>>>(Qb, Kb, Vb, mask, Ob);

    dim3 gOut(64, 16);
    mha_out_proj_kernel<<<gOut, 256, 0, stream>>>(Ob, Wo, bo, out);
}
